// IFHMM_64037962384098
// MI455X (gfx1250) — matrix operand order repaired and checked by reading, not compiled, not run on hardware
//
#include <hip/hip_runtime.h>
#include <hip/hip_bf16.h>
#include <hip/hip_fp16.h>

typedef __attribute__((ext_vector_type(16))) _Float16 v16h;
typedef __attribute__((ext_vector_type(8)))  float    v8f;

#define S_DIM 128
#define C_DIM 32

union FragU { v16h v; uint4 q[2]; };

// A-matrix 16x32 f16 fragment: lane half hi=0: K = kb+0..15;
// hi=1: K = kb+16..31 (contiguous). Row-major source row (16B aligned).
__device__ __forceinline__ v16h load_fragA(const _Float16* rowBase, int kb, int hi) {
    FragU f;
    const uint4* p = (const uint4*)(rowBase + kb + hi * 16);
    f.q[0] = p[0];   // halves 0..7
    f.q[1] = p[1];   // halves 8..15 (K +8)
    return f.v;
}
// B-matrix 32x16 f16 fragment: lane = column; hi=0: K = kb+0..15; hi=1: K = kb+16..31
// (contiguous). Source is the transposed matrix row (column of B), 16B aligned.
__device__ __forceinline__ v16h load_fragB(const _Float16* rowBase, int kb, int hi) {
    FragU f;
    const uint4* p = (const uint4*)(rowBase + kb + hi * 16);
    f.q[0] = p[0];
    f.q[1] = p[1];
    return f.v;
}

__device__ __forceinline__ v8f wmma16(v16h a, v16h b, v8f c) {
    return __builtin_amdgcn_wmma_f32_16x16x32_f16(false, a, false, b, (short)0, c,
                                                  false, false);
}

// ---------------- Kernel 1: emission row-softmax ([S,N] over N) ----------------
__global__ void __launch_bounds__(256)
emission_softmax_kernel(const float* __restrict__ em_un,
                        float* __restrict__ log_em_out,   // out4 [S,N]
                        _Float16* __restrict__ emT,       // ws [N,S] probs f16
                        int N) {
    const int s = blockIdx.x;
    const int tid = threadIdx.x;
    const float* row = em_un + (size_t)s * N;
    __shared__ float red[256];

    float m = -3.4e38f;
    for (int n = tid; n < N; n += 256) m = fmaxf(m, row[n]);
    red[tid] = m; __syncthreads();
    for (int off = 128; off > 0; off >>= 1) {
        if (tid < off) red[tid] = fmaxf(red[tid], red[tid + off]);
        __syncthreads();
    }
    m = red[0]; __syncthreads();

    float sum = 0.f;
    for (int n = tid; n < N; n += 256) sum += __expf(row[n] - m);
    red[tid] = sum; __syncthreads();
    for (int off = 128; off > 0; off >>= 1) {
        if (tid < off) red[tid] += red[tid + off];
        __syncthreads();
    }
    const float logden = m + __logf(red[0]);

    for (int n = tid; n < N; n += 256) {
        float l = row[n] - logden;
        log_em_out[(size_t)s * N + n] = l;
        emT[(size_t)n * S_DIM + s] = (_Float16)__expf(l);
    }
}

// ---------------- Kernel 2: chain-weight softmax ([T,C] over C=32, wave/row) ----
__global__ void __launch_bounds__(256)
chainw_softmax_kernel(const float* __restrict__ cw_un,
                      float* __restrict__ log_cw_out,  // out5 [T,C]
                      float* __restrict__ Wprob,       // ws [T,C] probs f32
                      int T) {
    const int w = (int)(blockIdx.x * blockDim.x + threadIdx.x) >> 5;
    const int lane = threadIdx.x & 31;
    if (w >= T) return;
    float v = cw_un[(size_t)w * C_DIM + lane];
    float m = v;
    for (int off = 16; off > 0; off >>= 1) m = fmaxf(m, __shfl_xor(m, off, 32));
    float sum = __expf(v - m);
    for (int off = 16; off > 0; off >>= 1) sum += __shfl_xor(sum, off, 32);
    float l = v - m - __logf(sum);
    log_cw_out[(size_t)w * C_DIM + lane] = l;
    Wprob[(size_t)w * C_DIM + lane] = __expf(l);
}

// ---------------- Kernel 3: init + transition softmax (row of 128) -------------
__global__ void __launch_bounds__(128)
init_trans_softmax_kernel(const float* __restrict__ init_un,   // [32,128]
                          const float* __restrict__ trans_un,  // [128,128]
                          float* __restrict__ log_h0_out,      // out3 t=0 [32,128]
                          _Float16* __restrict__ H_all,        // ws [T,32,128]; t=0
                          _Float16* __restrict__ TrT) {        // ws [128,128] T-posed
    const int r = blockIdx.x, tid = threadIdx.x;
    const bool isInit = r < C_DIM;
    const float* row = isInit ? (init_un + (size_t)r * S_DIM)
                              : (trans_un + (size_t)(r - C_DIM) * S_DIM);
    __shared__ float red[128];
    float v = row[tid];
    red[tid] = v; __syncthreads();
    for (int off = 64; off > 0; off >>= 1) {
        if (tid < off) red[tid] = fmaxf(red[tid], red[tid + off]);
        __syncthreads();
    }
    float m = red[0]; __syncthreads();
    red[tid] = __expf(v - m); __syncthreads();
    for (int off = 64; off > 0; off >>= 1) {
        if (tid < off) red[tid] += red[tid + off];
        __syncthreads();
    }
    float l = v - m - __logf(red[0]);
    if (isInit) {
        log_h0_out[(size_t)r * S_DIM + tid] = l;
        H_all[(size_t)r * S_DIM + tid] = (_Float16)__expf(l);
    } else {
        TrT[(size_t)tid * S_DIM + (r - C_DIM)] = (_Float16)__expf(l);
    }
}

// ---------------- Kernel 4: sequential chain scan H_t = H_{t-1} @ Tr -----------
// 1 block, 8 waves; Tr B-fragments pinned in registers across all steps.
__global__ void __launch_bounds__(256)
chain_scan_kernel(const _Float16* __restrict__ TrT,   // ws [128,128]
                  _Float16* __restrict__ H_all,       // ws [T,32,128] (t=0 filled)
                  float* __restrict__ log_h_out,      // out3 [T,32,128]
                  int T) {
    __shared__ __align__(16) _Float16 Hbuf[2][C_DIM][S_DIM];
    const int tid = threadIdx.x;
    const int wave = tid >> 5;          // 0..7 -> output state tile
    const int lane = tid & 31;
    const int lanelo = lane & 15;
    const int hi = lane >> 4;

    // persistent B fragments: columns wave*16+lanelo of Tr = rows of TrT
    const _Float16* rb = TrT + (size_t)(wave * 16 + lanelo) * S_DIM;
    const v16h b0 = load_fragB(rb, 0, hi);
    const v16h b1 = load_fragB(rb, 32, hi);
    const v16h b2 = load_fragB(rb, 64, hi);
    const v16h b3 = load_fragB(rb, 96, hi);

    // stage H0 into LDS (4096 halves = 512 uint4)
    {
        const uint4* src = (const uint4*)H_all;
        uint4* dst = (uint4*)&Hbuf[0][0][0];
        dst[tid] = src[tid];
        dst[tid + 256] = src[tid + 256];
    }
    __syncthreads();

    const int ncol = wave * 16 + lanelo;
    for (int t = 1; t < T; ++t) {
        const int cur = (t - 1) & 1, nxt = t & 1;
        const _Float16* rowA0 = &Hbuf[cur][lanelo][0];
        const _Float16* rowA1 = &Hbuf[cur][16 + lanelo][0];
        v8f acc0 = {}, acc1 = {};
        acc0 = wmma16(load_fragA(rowA0,  0, hi), b0, acc0);
        acc0 = wmma16(load_fragA(rowA0, 32, hi), b1, acc0);
        acc0 = wmma16(load_fragA(rowA0, 64, hi), b2, acc0);
        acc0 = wmma16(load_fragA(rowA0, 96, hi), b3, acc0);
        acc1 = wmma16(load_fragA(rowA1,  0, hi), b0, acc1);
        acc1 = wmma16(load_fragA(rowA1, 32, hi), b1, acc1);
        acc1 = wmma16(load_fragA(rowA1, 64, hi), b2, acc1);
        acc1 = wmma16(load_fragA(rowA1, 96, hi), b3, acc1);
        #pragma unroll
        for (int r = 0; r < 8; ++r) {
            const int c0 = r + hi * 8, c1 = 16 + c0;
            const float v0 = acc0[r], v1 = acc1[r];
            Hbuf[nxt][c0][ncol] = (_Float16)v0;
            Hbuf[nxt][c1][ncol] = (_Float16)v1;
            const size_t i0 = ((size_t)t * C_DIM + c0) * S_DIM + ncol;
            const size_t i1 = ((size_t)t * C_DIM + c1) * S_DIM + ncol;
            H_all[i0] = (_Float16)v0;
            H_all[i1] = (_Float16)v1;
            log_h_out[i0] = __logf(v0);
            log_h_out[i1] = __logf(v1);
        }
        __syncthreads();
    }
}

// ---------------- Kernel 5: O_t = H_t @ Em, fused log + chain-weight reduce ----
// grid = T * 4 blocks; block = 8 waves; each wave: 8 N-tiles of 16.
__global__ void __launch_bounds__(256)
emit_output_kernel(const _Float16* __restrict__ H_all,  // ws [T,32,128]
                   const _Float16* __restrict__ EmT,    // ws [N,128]
                   const float* __restrict__ Wprob,     // ws [T,32]
                   float* __restrict__ out_obs,         // out1 [T,N]
                   float* __restrict__ out_obs_,        // out2 [T,32,N]
                   int N) {
    const int t = blockIdx.x >> 2;
    const int chunk = blockIdx.x & 3;
    const int tid = threadIdx.x;
    const int wave = tid >> 5, lane = tid & 31;
    const int lanelo = lane & 15, hi = lane >> 4;

    __shared__ float Wlds[C_DIM];
    if (tid < C_DIM) Wlds[tid] = Wprob[(size_t)t * C_DIM + tid];
    __syncthreads();

    float wc0[8], wc1[8];
    #pragma unroll
    for (int r = 0; r < 8; ++r) {
        wc0[r] = Wlds[r + hi * 8];
        wc1[r] = Wlds[16 + r + hi * 8];
    }

    // A fragments for H_t: reused across all tiles handled by this wave
    const _Float16* rowA0 = H_all + ((size_t)t * C_DIM + lanelo) * S_DIM;
    const _Float16* rowA1 = H_all + ((size_t)t * C_DIM + 16 + lanelo) * S_DIM;
    const v16h a00 = load_fragA(rowA0,  0, hi), a01 = load_fragA(rowA0, 32, hi);
    const v16h a02 = load_fragA(rowA0, 64, hi), a03 = load_fragA(rowA0, 96, hi);
    const v16h a10 = load_fragA(rowA1,  0, hi), a11 = load_fragA(rowA1, 32, hi);
    const v16h a12 = load_fragA(rowA1, 64, hi), a13 = load_fragA(rowA1, 96, hi);

    const int tile0 = chunk * 64 + wave * 8;
    for (int i = 0; i < 8; ++i) {
        const int nt = tile0 + i;
        const int ncol = nt * 16 + lanelo;
        const _Float16* rbN = EmT + (size_t)ncol * S_DIM;
        const v16h b0 = load_fragB(rbN,  0, hi);
        const v16h b1 = load_fragB(rbN, 32, hi);
        const v16h b2 = load_fragB(rbN, 64, hi);
        const v16h b3 = load_fragB(rbN, 96, hi);
        v8f acc0 = {}, acc1 = {};
        acc0 = wmma16(a00, b0, acc0);
        acc0 = wmma16(a01, b1, acc0);
        acc0 = wmma16(a02, b2, acc0);
        acc0 = wmma16(a03, b3, acc0);
        acc1 = wmma16(a10, b0, acc1);
        acc1 = wmma16(a11, b1, acc1);
        acc1 = wmma16(a12, b2, acc1);
        acc1 = wmma16(a13, b3, acc1);

        float partial = 0.f;
        #pragma unroll
        for (int r = 0; r < 8; ++r) {
            const float v0 = acc0[r], v1 = acc1[r];
            partial += wc0[r] * v0 + wc1[r] * v1;
            const int c0 = r + hi * 8, c1 = 16 + c0;
            out_obs_[((size_t)t * C_DIM + c0) * N + ncol] = __logf(v0);
            out_obs_[((size_t)t * C_DIM + c1) * N + ncol] = __logf(v1);
        }
        partial += __shfl_xor(partial, 16, 32);
        if (hi == 0) out_obs[(size_t)t * N + ncol] = __logf(partial);
    }
}

extern "C" void kernel_launch(void* const* d_in, const int* in_sizes, int n_in,
                              void* d_out, int out_size, void* d_ws, size_t ws_size,
                              hipStream_t stream) {
    (void)n_in; (void)out_size; (void)ws_size;
    const float* init_un = (const float*)d_in[0];   // [32,128]
    const float* cw_un   = (const float*)d_in[1];   // [T,32]
    const float* em_un   = (const float*)d_in[2];   // [128,N]
    const float* tr_un   = (const float*)d_in[3];   // [128,128]
    const int T = in_sizes[1] / C_DIM;              // 512
    const int N = in_sizes[2] / S_DIM;              // 4096

    float* out = (float*)d_out;
    const size_t o1 = 0;                                   // [T,N]
    const size_t o2 = o1 + (size_t)T * N;                  // [T,C,N]
    const size_t o3 = o2 + (size_t)T * C_DIM * N;          // [T,C,S]
    const size_t o4 = o3 + (size_t)T * C_DIM * S_DIM;      // [S,N]
    const size_t o5 = o4 + (size_t)S_DIM * N;              // [T,C]

    char* ws = (char*)d_ws;
    _Float16* EmT   = (_Float16*)ws;  ws += (size_t)N * S_DIM * sizeof(_Float16);
    _Float16* TrT   = (_Float16*)ws;  ws += (size_t)S_DIM * S_DIM * sizeof(_Float16);
    _Float16* H_all = (_Float16*)ws;  ws += (size_t)T * C_DIM * S_DIM * sizeof(_Float16);
    float*    Wprob = (float*)ws;     ws += (size_t)T * C_DIM * sizeof(float);

    emission_softmax_kernel<<<S_DIM, 256, 0, stream>>>(em_un, out + o4, EmT, N);
    chainw_softmax_kernel<<<(T + 7) / 8, 256, 0, stream>>>(cw_un, out + o5, Wprob, T);
    init_trans_softmax_kernel<<<C_DIM + S_DIM, 128, 0, stream>>>(
        init_un, tr_un, out + o3, H_all, TrT);
    chain_scan_kernel<<<1, 256, 0, stream>>>(TrT, H_all, out + o3, T);
    emit_output_kernel<<<T * 4, 256, 0, stream>>>(H_all, EmT, Wprob,
                                                  out + o1, out + o2, N);
}
